// QANet_34213709480323
// MI455X (gfx1250) — matrix operand order repaired and checked by reading, not compiled, not run on hardware
//
#include <hip/hip_runtime.h>
#include <math.h>

// ---------------- model dimensions ----------------
#define BATCH 16
#define DIM   128
#define NH    8
#define DHEAD 16
#define LCW   400
#define LQW   50
#define CLW   16
#define WDW   300
#define CDW   64

typedef __attribute__((ext_vector_type(16))) _Float16 v16h;
typedef __attribute__((ext_vector_type(8)))  float    v8f;

#define GF_RELU    1
#define GF_TRANS_A 2
#define GF_TRANS_X 4

#define TK 32

static inline unsigned nblk(long n, int t) { return (unsigned)((n + t - 1) / t); }

// ============================================================
// Generic batched GEMM on the CDNA5 WMMA pipe.
//   Y[bz] = alpha * A[bz] (MxK) * X[bz] (KxN) (+bias[m]) (relu) (+R[bz])
// One wave (32 lanes) per 16x16 output tile; K stepped by 32.
// A/B tiles are staged through LDS as f16, fragments are built per the
// CDNA5 16-bit A(16x32)/B(32x16) VGPR layouts (ISA 7.12.2, wave32).
// ============================================================
__global__ __launch_bounds__(32)
void wmma_gemm_kernel(const float* __restrict__ A, long sAz, int ldA,
                      const float* __restrict__ X, long sXz, int xMod, int ldX,
                      float* __restrict__ Y, long sYz, int ldY,
                      const float* __restrict__ bias,
                      const float* __restrict__ R, long sRz,
                      int M, int N, int K, float alpha, int flags)
{
    __shared__ _Float16 As[16 * TK];   // [m][k]
    __shared__ _Float16 Bs[TK * 16];   // [k][n]

    const int bz    = blockIdx.z;
    const float* Ab = A + sAz * (long)bz;
    const float* Xb = X + sXz * (long)(xMod ? (bz % xMod) : bz);
    float*       Yb = Y + sYz * (long)bz;
    const float* Rb = R ? (R + sRz * (long)bz) : (const float*)0;

    const int m0   = blockIdx.y * 16;
    const int n0   = blockIdx.x * 16;
    const int lane = threadIdx.x;

    v8f acc = {};

    for (int k0 = 0; k0 < K; k0 += TK) {
        // Cooperative stage of A tile (16 x 32), f32 -> f16.
        for (int i = lane; i < 16 * TK; i += 32) {
            int m = i >> 5, k = i & (TK - 1);
            int gm = m0 + m, gk = k0 + k;
            float val = 0.f;
            if (gm < M && gk < K)
                val = (flags & GF_TRANS_A) ? Ab[(long)gk * ldA + gm]
                                           : Ab[(long)gm * ldA + gk];
            As[i] = (_Float16)val;
        }
        // Cooperative stage of B tile (32 x 16), f32 -> f16.
        for (int i = lane; i < TK * 16; i += 32) {
            int k = i >> 4, n = i & 15;
            int gk = k0 + k, gn = n0 + n;
            float val = 0.f;
            if (gk < K && gn < N)
                val = (flags & GF_TRANS_X) ? Xb[(long)gn * ldX + gk]
                                           : Xb[(long)gk * ldX + gn];
            Bs[i] = (_Float16)val;
        }
        __syncthreads();

        // Hint-prefetch the next K tiles (global_prefetch_b8).
        if (k0 + TK < K) {
            __builtin_prefetch(Ab + (long)m0 * ldA + (k0 + TK), 0, 1);
            __builtin_prefetch(Xb + (long)(k0 + TK) * ldX + n0, 0, 1);
        }

        // Build fragments per CDNA5 wave32 16-bit layouts.
        v16h af, bf;
        const int mrow = lane & 15;   // M row for A frag / N col for B frag
        const int grp  = lane >> 4;   // lane half-group
#pragma unroll
        for (int e = 0; e < 16; ++e) {
            int vv = e >> 1, hh = e & 1;
            // A 16x32: VGPR 0-3 -> K = grp*8 + 2v+h ; VGPR 4-7 -> K = 16 + grp*8 + 2(v-4)+h
            int ka = (vv < 4) ? (grp * 8 + vv * 2 + hh)
                              : (16 + grp * 8 + (vv - 4) * 2 + hh);
            af[e] = As[mrow * TK + ka];
            // B 32x16: same K order as A (e<8 -> grp*8+e ; e>=8 -> 16+grp*8+(e-8))
            bf[e] = Bs[ka * 16 + mrow];
        }
        acc = __builtin_amdgcn_wmma_f32_16x16x32_f16(false, af, false, bf,
                                                     (short)0, acc, false, false);
        __syncthreads();
    }

    // D layout: lane -> N = lane&15 ; VGPR r -> M = r + 8*(lane>>4)
    const int nc   = n0 + (lane & 15);
    const int grp2 = lane >> 4;
    if (nc < N) {
#pragma unroll
        for (int r = 0; r < 8; ++r) {
            int mr = m0 + r + grp2 * 8;
            if (mr < M) {
                float v = acc[r] * alpha;
                if (bias) v += bias[mr];
                if (flags & GF_RELU) v = v > 0.f ? v : 0.f;
                if (Rb) v += Rb[(long)mr * ldY + nc];
                Yb[(long)mr * ldY + nc] = v;
            }
        }
    }
}

// ============================================================
// Support kernels (wave32-shaped where it matters)
// ============================================================

__global__ void pos_enc_kernel(float* __restrict__ pe, int L)
{
    int idx = blockIdx.x * blockDim.x + threadIdx.x;
    if (idx >= DIM * L) return;
    int d = idx / L, l = idx % L;
    float freq, phase;
    if ((d & 1) == 0) { freq =  powf(10000.f, -(float)d       / (float)DIM); phase = 0.f; }
    else              { freq = -powf(10000.f, -(float)(d - 1) / (float)DIM); phase = 1.5707963267948966f; }
    pe[idx] = sinf((float)l * freq + phase);
}

// depthwise 5x5 conv over (L, CL) with on-the-fly char-embedding gather
__global__ void char_dw_kernel(const float* __restrict__ cm, const int* __restrict__ cid,
                               const float* __restrict__ w, const float* __restrict__ bsc,
                               float* __restrict__ out, int L, long n)
{
    long idx = (long)blockIdx.x * blockDim.x + threadIdx.x;
    if (idx >= n) return;
    int t = (int)(idx % CLW);
    int l = (int)((idx / CLW) % L);
    int c = (int)((idx / ((long)CLW * L)) % CDW);
    int b = (int)(idx / ((long)CLW * L * CDW));
    float acc = bsc[c];
    for (int dl = 0; dl < 5; ++dl) {
        int ll = l + dl - 2;
        if (ll < 0 || ll >= L) continue;
        for (int dt = 0; dt < 5; ++dt) {
            int tt = t + dt - 2;
            if (tt < 0 || tt >= CLW) continue;
            int id = cid[((long)b * L + ll) * CLW + tt];
            acc += cm[(long)id * CDW + c] * w[c * 25 + dl * 5 + dt];
        }
    }
    out[idx] = acc;
}

// pointwise CD->D + bias, relu, max over CL; writes channels [0,DIM) of e_cat
__global__ void char_pw_max_kernel(const float* __restrict__ pw, const float* __restrict__ pwb,
                                   const float* __restrict__ xd, float* __restrict__ ecat,
                                   int L, long n)
{
    long idx = (long)blockIdx.x * blockDim.x + threadIdx.x;
    if (idx >= n) return;
    int l = (int)(idx % L);
    int d = (int)((idx / L) % DIM);
    int b = (int)(idx / ((long)L * DIM));
    float best = 0.f;                       // max_t relu(v_t) == max(0, max_t v_t)
    for (int t = 0; t < CLW; ++t) {
        float acc = pwb[d];
        for (int c = 0; c < CDW; ++c)
            acc += pw[d * CDW + c] * xd[(((long)b * CDW + c) * L + l) * CLW + t];
        best = fmaxf(best, acc);
    }
    ecat[((long)b * (WDW + DIM) + d) * L + l] = best;
}

// word embedding gather into channels [DIM, DIM+WD) of e_cat
__global__ void word_gather_kernel(const float* __restrict__ wm, const int* __restrict__ wid,
                                   float* __restrict__ ecat, int L, long n)
{
    long idx = (long)blockIdx.x * blockDim.x + threadIdx.x;
    if (idx >= n) return;
    int l  = (int)(idx % L);
    int wd = (int)((idx / L) % WDW);
    int b  = (int)(idx / ((long)L * WDW));
    ecat[((long)b * (WDW + DIM) + DIM + wd) * L + l] =
        wm[(long)wid[(long)b * L + l] * WDW + wd];
}

// depthwise 1D conv, pad k/2, optional bias
__global__ void dwconv1d_kernel(const float* __restrict__ x, const float* __restrict__ w,
                                const float* __restrict__ bias, float* __restrict__ y,
                                int C, int L, int k, long n)
{
    long idx = (long)blockIdx.x * blockDim.x + threadIdx.x;
    if (idx >= n) return;
    int l = (int)(idx % L);
    int c = (int)((idx / L) % C);
    long rowbase = idx - l;
    int p = k / 2;
    float acc = bias ? bias[c] : 0.f;
    for (int j = 0; j < k; ++j) {
        int ll = l + j - p;
        if (ll >= 0 && ll < L) acc += x[rowbase + ll] * w[c * k + j];
    }
    y[idx] = acc;
}

// normalize each (b,d) row over L, ddof=1; one wave per row
__global__ __launch_bounds__(32)
void norm_rows_kernel(const float* __restrict__ x, float* __restrict__ y, int L)
{
    long row = blockIdx.x;
    const float* xr = x + row * (long)L;
    float* yr = y + row * (long)L;
    int lane = threadIdx.x;
    float s = 0.f;
    for (int i = lane; i < L; i += 32) s += xr[i];
    for (int o = 16; o > 0; o >>= 1) s += __shfl_xor(s, o, 32);
    float m = s / (float)L;
    float v = 0.f;
    for (int i = lane; i < L; i += 32) { float d = xr[i] - m; v += d * d; }
    for (int o = 16; o > 0; o >>= 1) v += __shfl_xor(v, o, 32);
    float sd = sqrtf(v / (float)(L - 1)) + 1e-6f;
    for (int i = lane; i < L; i += 32) yr[i] = (xr[i] - m) / sd;
}

__global__ void add_pe_kernel(float* __restrict__ x, const float* __restrict__ pe,
                              long n, int DL)
{
    long idx = (long)blockIdx.x * blockDim.x + threadIdx.x;
    if (idx >= n) return;
    x[idx] += pe[idx % DL];
}

__global__ void highway_combine_kernel(const float* __restrict__ g, const float* __restrict__ nl,
                                       const float* __restrict__ h, float* __restrict__ out, long n)
{
    long idx = (long)blockIdx.x * blockDim.x + threadIdx.x;
    if (idx >= n) return;
    float gg = 1.f / (1.f + expf(-g[idx]));
    out[idx] = gg * nl[idx] + (1.f - gg) * h[idx];
}

// softmax over the last dim; one wave per row
__global__ __launch_bounds__(32)
void softmax_rows_kernel(const float* __restrict__ x, float* __restrict__ y, int N)
{
    long row = blockIdx.x;
    const float* xr = x + row * (long)N;
    float* yr = y + row * (long)N;
    int lane = threadIdx.x;
    float mx = -3.4e38f;
    for (int i = lane; i < N; i += 32) mx = fmaxf(mx, xr[i]);
    for (int o = 16; o > 0; o >>= 1) mx = fmaxf(mx, __shfl_xor(mx, o, 32));
    float sum = 0.f;
    for (int i = lane; i < N; i += 32) sum += expf(xr[i] - mx);
    for (int o = 16; o > 0; o >>= 1) sum += __shfl_xor(sum, o, 32);
    float inv = 1.f / sum;
    for (int i = lane; i < N; i += 32) yr[i] = expf(xr[i] - mx) * inv;
}

// softmax down columns of an (R x Ncol) matrix (batch = blockIdx.z); wave per column
__global__ __launch_bounds__(32)
void softmax_cols_kernel(const float* __restrict__ x, float* __restrict__ y, int R, int Ncol)
{
    long base = (long)blockIdx.z * R * Ncol;
    int j = blockIdx.x;
    int lane = threadIdx.x;
    const float* xb = x + base;
    float* yb = y + base;
    float mx = -3.4e38f;
    for (int i = lane; i < R; i += 32) mx = fmaxf(mx, xb[(long)i * Ncol + j]);
    for (int o = 16; o > 0; o >>= 1) mx = fmaxf(mx, __shfl_xor(mx, o, 32));
    float sum = 0.f;
    for (int i = lane; i < R; i += 32) sum += expf(xb[(long)i * Ncol + j] - mx);
    for (int o = 16; o > 0; o >>= 1) sum += __shfl_xor(sum, o, 32);
    float inv = 1.f / sum;
    for (int i = lane; i < R; i += 32)
        yb[(long)i * Ncol + j] = expf(xb[(long)i * Ncol + j] - mx) * inv;
}

// heads (h,b,dk,l) -> head (b, h*DHEAD+dk, l)
__global__ void heads_transpose_kernel(const float* __restrict__ heads, float* __restrict__ out,
                                       int L, long n)
{
    long idx = (long)blockIdx.x * blockDim.x + threadIdx.x;
    if (idx >= n) return;
    int l = (int)(idx % L);
    int r = (int)((idx / L) % DHEAD);
    int b = (int)((idx / ((long)L * DHEAD)) % BATCH);
    int h = (int)(idx / ((long)L * DHEAD * BATCH));
    out[((long)b * DIM + h * DHEAD + r) * L + l] = heads[idx];
}

// Cm[b,d,i] = C[b,d,i] * cqW[b, 2D + d]
__global__ void scale_mul_kernel(const float* __restrict__ Cin, const float* __restrict__ W,
                                 float* __restrict__ out, int L, long n)
{
    long idx = (long)blockIdx.x * blockDim.x + threadIdx.x;
    if (idx >= n) return;
    int d = (int)((idx / L) % DIM);
    int b = (int)(idx / ((long)L * DIM));
    out[idx] = Cin[idx] * W[(long)b * 3 * DIM + 2 * DIM + d];
}

// out[b,l] = sum_d W[b, woff + d] * x[b,d,l]
__global__ void wsum_kernel(const float* __restrict__ x, const float* __restrict__ W,
                            int woff, float* __restrict__ out, int L, long n)
{
    long idx = (long)blockIdx.x * blockDim.x + threadIdx.x;
    if (idx >= n) return;
    int l = (int)(idx % L);
    int b = (int)(idx / L);
    float acc = 0.f;
    for (int d = 0; d < DIM; ++d)
        acc += W[(long)b * 3 * DIM + woff + d] * x[((long)b * DIM + d) * L + l];
    out[idx] = acc;
}

__global__ void cq_addterms_kernel(float* __restrict__ S, const float* __restrict__ ct,
                                   const float* __restrict__ qt, long n)
{
    long idx = (long)blockIdx.x * blockDim.x + threadIdx.x;
    if (idx >= n) return;
    int j = (int)(idx % LQW);
    int i = (int)((idx / LQW) % LCW);
    int b = (int)(idx / ((long)LQW * LCW));
    S[idx] += ct[(long)b * LCW + i] + qt[(long)b * LQW + j];
}

// X = concat([C, A, C*A, C*Bm], channel axis) -> (B, 4D, LC)
__global__ void cq_assemble_kernel(const float* __restrict__ Cin, const float* __restrict__ Aat,
                                   const float* __restrict__ Bm, float* __restrict__ X, long n)
{
    long idx = (long)blockIdx.x * blockDim.x + threadIdx.x;
    if (idx >= n) return;
    int i = (int)(idx % LCW);
    int d = (int)((idx / LCW) % DIM);
    int b = (int)(idx / ((long)LCW * DIM));
    float c = Cin[idx], a = Aat[idx], bm = Bm[idx];
    long base = (long)b * 4 * DIM * LCW;
    X[base + (long)d * LCW + i]             = c;
    X[base + (long)(DIM + d) * LCW + i]     = a;
    X[base + (long)(2 * DIM + d) * LCW + i] = c * a;
    X[base + (long)(3 * DIM + d) * LCW + i] = c * bm;
}

// Y[b,l] = W[b,:2D] . concat(Ma, Mb)[b,:,l]
__global__ void final_logits_kernel(const float* __restrict__ W, const float* __restrict__ Ma,
                                    const float* __restrict__ Mb, float* __restrict__ out,
                                    int L, long n)
{
    long idx = (long)blockIdx.x * blockDim.x + threadIdx.x;
    if (idx >= n) return;
    int l = (int)(idx % L);
    int b = (int)(idx / L);
    const float* wr = W + (long)b * 2 * DIM;
    float acc = 0.f;
    for (int d = 0; d < DIM; ++d) {
        acc += wr[d]       * Ma[((long)b * DIM + d) * L + l];
        acc += wr[DIM + d] * Mb[((long)b * DIM + d) * L + l];
    }
    out[idx] = acc;
}

// ============================================================
// Host-side orchestration
// ============================================================

static void launch_gemm(hipStream_t s,
                        const float* A, long sAz, int ldA,
                        const float* X, long sXz, int xMod, int ldX,
                        float* Y, long sYz, int ldY,
                        const float* bias, const float* R, long sRz,
                        int M, int N, int K, int nb, float alpha, int flags)
{
    dim3 grid((N + 15) / 16, (M + 15) / 16, nb);
    wmma_gemm_kernel<<<grid, dim3(32), 0, s>>>(A, sAz, ldA, X, sXz, xMod, ldX,
                                               Y, sYz, ldY, bias, R, sRz,
                                               M, N, K, alpha, flags);
}

struct Enc { const float *dw, *pw, *Wq, *Wk, *Wv, *Wo, *W; int nconv, k; };

static Enc make_enc(void* const* din, int b0, int nconv, int k)
{
    Enc e;
    e.dw = (const float*)din[b0 + 0]; e.pw = (const float*)din[b0 + 1];
    e.Wq = (const float*)din[b0 + 2]; e.Wk = (const float*)din[b0 + 3];
    e.Wv = (const float*)din[b0 + 4]; e.Wo = (const float*)din[b0 + 5];
    e.W  = (const float*)din[b0 + 6];
    e.nconv = nconv; e.k = k;
    return e;
}

struct Scratch { float *t1, *t2, *qb, *kb, *vb, *sc, *hb, *ht; };

// char/word embedding + dsconv + 2x highway; result (B,DIM,L) in `out`
static void run_embedding(hipStream_t s, void* const* din, const int* wid, const int* cid,
                          int L, float* xd, float* ecat, float* out, float* alt,
                          float* g, float* nl)
{
    const float* word_mat = (const float*)din[4];
    const float* char_mat = (const float*)din[5];
    long n1 = (long)BATCH * CDW * L * CLW;
    char_dw_kernel<<<nblk(n1, 256), 256, 0, s>>>(char_mat, cid,
        (const float*)din[6], (const float*)din[7], xd, L, n1);
    long n2 = (long)BATCH * DIM * L;
    char_pw_max_kernel<<<nblk(n2, 256), 256, 0, s>>>(
        (const float*)din[8], (const float*)din[9], xd, ecat, L, n2);
    long n3 = (long)BATCH * WDW * L;
    word_gather_kernel<<<nblk(n3, 256), 256, 0, s>>>(word_mat, wid, ecat, L, n3);
    // depthwise over 428 channels (k=5) with bias, into xd (reused as temp)
    long n4 = (long)BATCH * (WDW + DIM) * L;
    dwconv1d_kernel<<<nblk(n4, 256), 256, 0, s>>>(ecat, (const float*)din[10],
        (const float*)din[11], xd, WDW + DIM, L, 5, n4);
    // pointwise 428 -> 128 + bias (WMMA)
    launch_gemm(s, (const float*)din[12], 0, WDW + DIM,
                xd, (long)(WDW + DIM) * L, 0, L,
                out, (long)DIM * L, L,
                (const float*)din[13], nullptr, 0,
                DIM, L, WDW + DIM, BATCH, 1.f, 0);
    // highway x2 (channel-dim GEMMs on the WMMA pipe)
    const float* lin_w  = (const float*)din[14];
    const float* lin_b  = (const float*)din[15];
    const float* gate_w = (const float*)din[16];
    const float* gate_b = (const float*)din[17];
    float* cur = out; float* oth = alt;
    long n5 = (long)BATCH * DIM * L;
    for (int i = 0; i < 2; ++i) {
        launch_gemm(s, gate_w + (long)i * DIM * DIM, 0, DIM, cur, (long)DIM * L, 0, L,
                    g, (long)DIM * L, L, gate_b + i * DIM, nullptr, 0,
                    DIM, L, DIM, BATCH, 1.f, 0);
        launch_gemm(s, lin_w + (long)i * DIM * DIM, 0, DIM, cur, (long)DIM * L, 0, L,
                    nl, (long)DIM * L, L, lin_b + i * DIM, nullptr, 0,
                    DIM, L, DIM, BATCH, 1.f, GF_RELU);
        highway_combine_kernel<<<nblk(n5, 256), 256, 0, s>>>(g, nl, cur, oth, n5);
        float* t = cur; cur = oth; oth = t;
    }
    // two swaps -> result is back in `out`
}

// One QANet encoder block; result stays in x (nconv+2 is even for all blocks)
static void run_encoder(hipStream_t s, const Enc& ep, float* x, float* alt,
                        const float* pe, int L, const Scratch& w)
{
    long nBDL = (long)BATCH * DIM * L;
    add_pe_kernel<<<nblk(nBDL, 256), 256, 0, s>>>(x, pe, nBDL, DIM * L);
    float* cur = x; float* nxt = alt;

    for (int i = 0; i < ep.nconv; ++i) {
        norm_rows_kernel<<<BATCH * DIM, 32, 0, s>>>(cur, w.t1, L);
        dwconv1d_kernel<<<nblk(nBDL, 256), 256, 0, s>>>(w.t1,
            ep.dw + (long)i * DIM * ep.k, nullptr, w.t2, DIM, L, ep.k, nBDL);
        launch_gemm(s, ep.pw + (long)i * DIM * DIM, 0, DIM, w.t2, (long)DIM * L, 0, L,
                    nxt, (long)DIM * L, L, nullptr, cur, (long)DIM * L,
                    DIM, L, DIM, BATCH, 1.f, GF_RELU);
        float* t = cur; cur = nxt; nxt = t;
    }

    norm_rows_kernel<<<BATCH * DIM, 32, 0, s>>>(cur, w.t1, L);
    // Q/K/V: batched over (h,b) = 128, x batch = bz % BATCH
    launch_gemm(s, ep.Wq, (long)DHEAD * DIM, DIM, w.t1, (long)DIM * L, BATCH, L,
                w.qb, (long)DHEAD * L, L, nullptr, nullptr, 0,
                DHEAD, L, DIM, NH * BATCH, 1.f, 0);
    launch_gemm(s, ep.Wk, (long)DHEAD * DIM, DIM, w.t1, (long)DIM * L, BATCH, L,
                w.kb, (long)DHEAD * L, L, nullptr, nullptr, 0,
                DHEAD, L, DIM, NH * BATCH, 1.f, 0);
    launch_gemm(s, ep.Wv, (long)DHEAD * DIM, DIM, w.t1, (long)DIM * L, BATCH, L,
                w.vb, (long)DHEAD * L, L, nullptr, nullptr, 0,
                DHEAD, L, DIM, NH * BATCH, 1.f, 0);
    // scores = (q^T k) / sqrt(DHEAD)   (transA: A(m=i,k) = q[k*L+i])
    launch_gemm(s, w.qb, (long)DHEAD * L, L, w.kb, (long)DHEAD * L, 0, L,
                w.sc, (long)L * L, L, nullptr, nullptr, 0,
                L, L, DHEAD, NH * BATCH, 0.25f, GF_TRANS_A);
    // softmax over i (down the columns), in place
    softmax_cols_kernel<<<dim3(L, 1, NH * BATCH), 32, 0, s>>>(w.sc, w.sc, L, L);
    // heads = v (DHEADxL) * a (LxL)
    launch_gemm(s, w.vb, (long)DHEAD * L, L, w.sc, (long)L * L, 0, L,
                w.hb, (long)DHEAD * L, L, nullptr, nullptr, 0,
                DHEAD, L, L, NH * BATCH, 1.f, 0);
    heads_transpose_kernel<<<nblk(nBDL, 256), 256, 0, s>>>(w.hb, w.ht, L, nBDL);
    // out = Wo[b] * head + residual
    launch_gemm(s, ep.Wo, (long)DIM * DIM, DIM, w.ht, (long)DIM * L, 0, L,
                nxt, (long)DIM * L, L, nullptr, cur, (long)DIM * L,
                DIM, L, DIM, BATCH, 1.f, 0);
    { float* t = cur; cur = nxt; nxt = t; }

    norm_rows_kernel<<<BATCH * DIM, 32, 0, s>>>(cur, w.t1, L);
    launch_gemm(s, ep.W, (long)DIM * DIM, DIM, w.t1, (long)DIM * L, 0, L,
                nxt, (long)DIM * L, L, nullptr, cur, (long)DIM * L,
                DIM, L, DIM, BATCH, 1.f, GF_RELU);
    { float* t = cur; cur = nxt; nxt = t; }
    // even number of swaps -> result in x
}

static void run_cq(hipStream_t s, void* const* din, const float* Cb, const float* Qb,
                   float* Cm, float* S, float* S1, float* S2, float* ct, float* qt,
                   float* Aat, float* Tm, float* Bm, float* Xc)
{
    const float* W = (const float*)din[32];   // cq_W (B,1,3D)
    long nC = (long)BATCH * DIM * LCW;
    scale_mul_kernel<<<nblk(nC, 256), 256, 0, s>>>(Cb, W, Cm, LCW, nC);
    // S = Cm^T (LCxD) * Q (DxLQ)
    launch_gemm(s, Cm, (long)DIM * LCW, LCW, Qb, (long)DIM * LQW, 0, LQW,
                S, (long)LCW * LQW, LQW, nullptr, nullptr, 0,
                LCW, LQW, DIM, BATCH, 1.f, GF_TRANS_A);
    long nci = (long)BATCH * LCW;
    wsum_kernel<<<nblk(nci, 256), 256, 0, s>>>(Cb, W, DIM, ct, LCW, nci);   // wc . C
    long nqj = (long)BATCH * LQW;
    wsum_kernel<<<nblk(nqj, 256), 256, 0, s>>>(Qb, W, 0, qt, LQW, nqj);     // wq . Q
    long nS = (long)BATCH * LCW * LQW;
    cq_addterms_kernel<<<nblk(nS, 256), 256, 0, s>>>(S, ct, qt, nS);
    softmax_rows_kernel<<<BATCH * LCW, 32, 0, s>>>(S, S1, LQW);             // over j
    softmax_cols_kernel<<<dim3(LQW, 1, BATCH), 32, 0, s>>>(S, S2, LCW, LQW); // over i
    // A = Q (DxLQ) * S1^T (LQxLC)   (transX: X(k=j,n=i) = S1[i*LQ+j])
    launch_gemm(s, Qb, (long)DIM * LQW, LQW, S1, (long)LCW * LQW, 0, LQW,
                Aat, (long)DIM * LCW, LCW, nullptr, nullptr, 0,
                DIM, LCW, LQW, BATCH, 1.f, GF_TRANS_X);
    // T = S1 (LCxLQ) * S2^T (LQxLC)
    launch_gemm(s, S1, (long)LCW * LQW, LQW, S2, (long)LCW * LQW, 0, LQW,
                Tm, (long)LCW * LCW, LCW, nullptr, nullptr, 0,
                LCW, LCW, LQW, BATCH, 1.f, GF_TRANS_X);
    // Bm = C (DxLC) * T^T (LCxLC)
    launch_gemm(s, Cb, (long)DIM * LCW, LCW, Tm, (long)LCW * LCW, 0, LCW,
                Bm, (long)DIM * LCW, LCW, nullptr, nullptr, 0,
                DIM, LCW, LCW, BATCH, 1.f, GF_TRANS_X);
    cq_assemble_kernel<<<nblk(nC, 256), 256, 0, s>>>(Cb, Aat, Bm, Xc, nC);
}

extern "C" void kernel_launch(void* const* d_in, const int* in_sizes, int n_in,
                              void* d_out, int out_size, void* d_ws, size_t ws_size,
                              hipStream_t stream)
{
    (void)in_sizes; (void)n_in; (void)out_size; (void)ws_size;
    // Inputs (setup_inputs dict order, params flattened depth-first):
    // 0 Cwid 1 Ccid 2 Qwid 3 Qcid 4 word_mat 5 char_mat
    // 6..17 emb{c2_dw,c2_dwb,c2_pw,c2_pwb,c1_dw,c1_dwb,c1_pw,c1_pwb,lin_w,lin_b,gate_w,gate_b}
    // 18..24 c_enc{dw,pw,Wq,Wk,Wv,Wo,W}  25..31 q_enc  32 cq_W  33 rz_dw  34 rz_pw
    // 35..41 m0  42..48 m1  49..55 m2  56 W0  57 W1
    const int* Cwid = (const int*)d_in[0];
    const int* Ccid = (const int*)d_in[1];
    const int* Qwid = (const int*)d_in[2];
    const int* Qcid = (const int*)d_in[3];
    float* out = (float*)d_out;

    // workspace bump allocator
    char* base = (char*)d_ws; size_t off = 0;
    auto alloc = [&](long n) -> float* {
        float* p = (float*)(base + off);
        off = (off + (size_t)n * sizeof(float) + 255) & ~(size_t)255;
        return p;
    };
    float* peC  = alloc((long)DIM * LCW);
    float* peQ  = alloc((long)DIM * LQW);
    float* xd   = alloc((long)BATCH * CDW * LCW * CLW);      // 6.55M; also dsconv temps
    float* ecat = alloc((long)BATCH * (WDW + DIM) * LCW);
    float* bufC = alloc((long)BATCH * DIM * LCW);
    float* altC = alloc((long)BATCH * DIM * LCW);
    float* bufQ = alloc((long)BATCH * DIM * LQW);
    float* altQ = alloc((long)BATCH * DIM * LQW);
    float* t1   = alloc((long)BATCH * DIM * LCW);
    float* t2   = alloc((long)BATCH * DIM * LCW);
    float* qb   = alloc((long)NH * BATCH * DHEAD * LCW);
    float* kb   = alloc((long)NH * BATCH * DHEAD * LCW);
    float* vb   = alloc((long)NH * BATCH * DHEAD * LCW);
    float* sc   = alloc((long)NH * BATCH * LCW * LCW);       // 20.5M (fits L2)
    float* hb   = alloc((long)NH * BATCH * DHEAD * LCW);
    float* ht   = alloc((long)BATCH * DIM * LCW);
    float* S    = alloc((long)BATCH * LCW * LQW);
    float* S1   = alloc((long)BATCH * LCW * LQW);
    float* S2   = alloc((long)BATCH * LCW * LQW);
    float* ct   = alloc((long)BATCH * LCW);
    float* qt   = alloc((long)BATCH * LQW);
    float* Cm   = alloc((long)BATCH * DIM * LCW);
    float* Aat  = alloc((long)BATCH * DIM * LCW);
    float* Tm   = alloc((long)BATCH * LCW * LCW);
    float* Bm   = alloc((long)BATCH * DIM * LCW);
    float* Xc   = alloc((long)BATCH * 4 * DIM * LCW);
    float* M0s  = alloc((long)BATCH * DIM * LCW);
    float* M1s  = alloc((long)BATCH * DIM * LCW);
    float* lg   = alloc((long)2 * BATCH * LCW);

    Scratch w; w.t1 = t1; w.t2 = t2; w.qb = qb; w.kb = kb; w.vb = vb;
    w.sc = sc; w.hb = hb; w.ht = ht;

    // positional encodings
    pos_enc_kernel<<<nblk((long)DIM * LCW, 256), 256, 0, stream>>>(peC, LCW);
    pos_enc_kernel<<<nblk((long)DIM * LQW, 256), 256, 0, stream>>>(peQ, LQW);

    // embeddings
    run_embedding(stream, d_in, Cwid, Ccid, LCW, xd, ecat, bufC, altC, t1, t2);
    run_embedding(stream, d_in, Qwid, Qcid, LQW, xd, ecat, bufQ, altQ, t1, t2);

    // context / question encoders
    Enc cenc = make_enc(d_in, 18, 4, 7);
    Enc qenc = make_enc(d_in, 25, 4, 7);
    run_encoder(stream, cenc, bufC, altC, peC, LCW, w);
    run_encoder(stream, qenc, bufQ, altQ, peQ, LQW, w);

    // context-query attention -> Xc (B, 4D, LC)
    run_cq(stream, d_in, bufC, bufQ, Cm, S, S1, S2, ct, qt, Aat, Tm, Bm, Xc);

    // resize dsconv: depthwise 512ch k=5 then pointwise 512->128
    long nrz = (long)BATCH * 4 * DIM * LCW;
    dwconv1d_kernel<<<nblk(nrz, 256), 256, 0, stream>>>(Xc, (const float*)d_in[33],
        nullptr, xd, 4 * DIM, LCW, 5, nrz);
    launch_gemm(stream, (const float*)d_in[34], 0, 4 * DIM,
                xd, (long)4 * DIM * LCW, 0, LCW,
                bufC, (long)DIM * LCW, LCW, nullptr, nullptr, 0,
                DIM, LCW, 4 * DIM, BATCH, 1.f, 0);

    // model encoder stacks
    Enc m0 = make_enc(d_in, 35, 2, 5);
    Enc m1 = make_enc(d_in, 42, 2, 5);
    Enc m2 = make_enc(d_in, 49, 2, 5);
    size_t mbytes = (size_t)BATCH * DIM * LCW * sizeof(float);
    for (int i = 0; i < 7; ++i) run_encoder(stream, m0, bufC, altC, peC, LCW, w);
    hipMemcpyAsync(M0s, bufC, mbytes, hipMemcpyDeviceToDevice, stream);
    for (int i = 0; i < 7; ++i) run_encoder(stream, m1, bufC, altC, peC, LCW, w);
    hipMemcpyAsync(M1s, bufC, mbytes, hipMemcpyDeviceToDevice, stream);
    for (int i = 0; i < 7; ++i) run_encoder(stream, m2, bufC, altC, peC, LCW, w);

    // output heads: Y0 from [M0,M1], Y1 from [M0,M2]; softmax over L
    long nl0 = (long)BATCH * LCW;
    final_logits_kernel<<<nblk(nl0, 256), 256, 0, stream>>>(
        (const float*)d_in[56], M0s, M1s, lg, LCW, nl0);
    softmax_rows_kernel<<<BATCH, 32, 0, stream>>>(lg, out, LCW);
    final_logits_kernel<<<nblk(nl0, 256), 256, 0, stream>>>(
        (const float*)d_in[57], M0s, bufC, lg + nl0, LCW, nl0);
    softmax_rows_kernel<<<BATCH, 32, 0, stream>>>(lg + nl0, out + nl0, LCW);
}
